// EdgeAwareGNN_31851477467868
// MI455X (gfx1250) — compile-verified, matrix operand order checked by reading, not run on hardware
//
#include <hip/hip_runtime.h>
#include <hip/hip_bf16.h>

typedef __attribute__((ext_vector_type(16))) __bf16 v16bf;
typedef __attribute__((ext_vector_type(8)))  float  v8f;

#define N_NODES 20000
#define N_EDGES 100000
#define HIDDEN  64
#define N_GRAPHS 64

// ---------------------------------------------------------------- utilities
__global__ void zero_kernel(float* __restrict__ p, int n) {
    int t = blockIdx.x * blockDim.x + threadIdx.x;
    if (t < n) p[t] = 0.0f;
}

__global__ void deg_kernel(const int* __restrict__ eidx, float* __restrict__ deg, int E) {
    int t = blockIdx.x * blockDim.x + threadIdx.x;
    if (t < E) atomicAdd(&deg[eidx[E + t]], 1.0f);   // row 1 = dst
}

__global__ void cnt_kernel(const int* __restrict__ batch, float* __restrict__ cnt, int n) {
    int t = blockIdx.x * blockDim.x + threadIdx.x;
    if (t < n) atomicAdd(&cnt[batch[t]], 1.0f);
}

__global__ void final_pool_kernel(const float* __restrict__ pool, const float* __restrict__ cnt,
                                  float* __restrict__ out) {
    int t = blockIdx.x * blockDim.x + threadIdx.x;
    if (t < N_GRAPHS * HIDDEN) out[t] = pool[t] / fmaxf(cnt[t >> 6], 1.0f);
}

// ------------------------------------------------- pack B' into WMMA B-fragment order
// B'[(i,k), o] = eW2[k, i*64+o], K' = INC*KH.  Element order:
// tid = ((chunk*4 + oc)*32 + lane)*16 + e ;  lane: half=lane>>4, n=lane&15
// K-in-chunk = e + 8*half + (e&8)   (mirrors documented 16-bit A layout)
template<int KH, int INC>
__global__ void pack_b_kernel(const float* __restrict__ eW2, __bf16* __restrict__ bpack) {
    constexpr int TOT = KH * INC * 64;
    constexpr int LSH = (KH == 128) ? 7 : 6;
    int tid = blockIdx.x * blockDim.x + threadIdx.x;
    if (tid >= TOT) return;
    int e    = tid & 15;
    int lane = (tid >> 4) & 31;
    int oc   = (tid >> 9) & 3;
    int c    = tid >> 11;
    int hf   = lane >> 4;
    int n    = lane & 15;
    int K    = e + 8 * hf + (e & 8);
    int g    = c * 32 + K;
    int i    = g >> LSH;
    int k    = g & (KH - 1);
    bpack[tid] = (__bf16)eW2[k * (INC * 64) + i * 64 + oc * 16 + n];
}

// ------------------------------------------------- per-node eb2 bias vector
// hb[n,o] = sum_i h[n,i] * eb2[i*64+o]   (one wave per node, 2 outputs/lane)
template<int INC>
__global__ __launch_bounds__(256) void hb_kernel(const float* __restrict__ h,
                                                 const float* __restrict__ eb2,
                                                 float* __restrict__ hb, int nNodes) {
    int wid = threadIdx.x >> 5, lane = threadIdx.x & 31;
    int nid = blockIdx.x * 8 + wid;
    if (nid >= nNodes) return;
    #pragma unroll
    for (int q = 0; q < 2; ++q) {
        int o = lane + 32 * q;
        float acc = 0.0f;
        for (int i = 0; i < INC; ++i) acc += h[nid * INC + i] * eb2[i * 64 + o];
        hb[nid * 64 + o] = acc;
    }
}

// ------------------------------------------------- fused edge kernel (the GEMM)
// One wave = 16 edges.  msg[16,64] = A'[16, INC*KH] x B'[INC*KH, 64]
// A'[t,(i,k)] = h[src_t, i] * relu(edge_attr_t @ eW1 + eb1)[k]  built on the fly.
template<int KH, int INC>
__global__ __launch_bounds__(32) void edge_msg_kernel(
    const float* __restrict__ h,      // [N, INC]
    const int*   __restrict__ eidx,   // [2, E]
    const float* __restrict__ ea,     // [E, 4]
    const float* __restrict__ eW1,    // [4, KH]
    const float* __restrict__ eb1,    // [KH]
    const __bf16* __restrict__ bpack, // prepacked B'
    const float* __restrict__ hb,     // [N, 64] per-node eb2 term
    float* __restrict__ agg,          // [N, 64] scatter target
    int E)
{
    __shared__ float aS[16][KH + 4];
    __shared__ float hS[16][INC + 4];
    __shared__ float hbS[16][64 + 4];
    __shared__ float eaS[16][8];
    __shared__ int   sI[16], dI[16];

    const int lane = threadIdx.x;
    const int e0 = blockIdx.x * 16;

    if (lane < 16) {
        int e = e0 + lane;
        int ec = (e < E) ? e : (E - 1);
        sI[lane] = eidx[ec];
        dI[lane] = (e < E) ? eidx[E + ec] : -1;
    }
    for (int idx = lane; idx < 64; idx += 32) {
        int t = idx >> 2, j = idx & 3;
        int e = e0 + t;
        eaS[t][j] = (e < E) ? ea[((e < E) ? e : 0) * 4 + j] : 0.0f;
    }
    __syncthreads();

    // edge MLP hidden: a = relu(ea @ eW1 + eb1)
    for (int idx = lane; idx < 16 * KH; idx += 32) {
        int t = idx & 15, k = idx >> 4;
        float acc = eb1[k];
        #pragma unroll
        for (int j = 0; j < 4; ++j) acc += eaS[t][j] * eW1[j * KH + k];
        aS[t][k] = fmaxf(acc, 0.0f);
    }
    // gather h[src] and hb[src]
    for (int idx = lane; idx < 16 * INC; idx += 32) {
        int t = idx / INC, i = idx % INC;
        hS[t][i] = h[sI[t] * INC + i];
    }
    for (int idx = lane; idx < 16 * 64; idx += 32) {
        int t = idx >> 6, o = idx & 63;
        hbS[t][o] = hb[sI[t] * 64 + o];
    }
    __syncthreads();

    const int m  = lane & 15;
    const int hf = lane >> 4;
    constexpr int LSH  = (KH == 128) ? 7 : 6;
    constexpr int NCHK = (KH * INC) / 32;

    v8f acc0 = {}, acc1 = {}, acc2 = {}, acc3 = {};

    for (int c = 0; c < NCHK; ++c) {
        const int g0 = c * 32;
        const int i  = g0 >> LSH;
        const int kb = g0 & (KH - 1);
        const float hv = hS[m][i];
        v16bf aF;
        #pragma unroll
        for (int e = 0; e < 16; ++e) {
            int kk = kb + e + 8 * hf + (e & 8);
            aF[e] = (__bf16)(hv * aS[m][kk]);
        }
        const __bf16* bp = bpack + (size_t)c * 2048 + lane * 16;
        v16bf b0 = *(const v16bf*)(bp);
        v16bf b1 = *(const v16bf*)(bp + 512);
        v16bf b2 = *(const v16bf*)(bp + 1024);
        v16bf b3 = *(const v16bf*)(bp + 1536);
        acc0 = __builtin_amdgcn_wmma_f32_16x16x32_bf16(false, aF, false, b0, (short)0, acc0, false, false);
        acc1 = __builtin_amdgcn_wmma_f32_16x16x32_bf16(false, aF, false, b1, (short)0, acc1, false, false);
        acc2 = __builtin_amdgcn_wmma_f32_16x16x32_bf16(false, aF, false, b2, (short)0, acc2, false, false);
        acc3 = __builtin_amdgcn_wmma_f32_16x16x32_bf16(false, aF, false, b3, (short)0, acc3, false, false);
    }

    // D layout: lane holds column n = lane&15; element j is row (j + 8*half)
    const int n = m;
    #pragma unroll
    for (int j = 0; j < 8; ++j) {
        int mr = j + 8 * hf;
        int d  = dI[mr];
        if (d >= 0) {
            atomicAdd(&agg[d * 64 +  0 + n], acc0[j] + hbS[mr][ 0 + n]);
            atomicAdd(&agg[d * 64 + 16 + n], acc1[j] + hbS[mr][16 + n]);
            atomicAdd(&agg[d * 64 + 32 + n], acc2[j] + hbS[mr][32 + n]);
            atomicAdd(&agg[d * 64 + 48 + n], acc3[j] + hbS[mr][48 + n]);
        }
    }
}

// ------------------------------------------------- node update: mean-agg + root + bias, relu, LN
template<int INC, bool FINAL>
__global__ __launch_bounds__(256) void node_update_kernel(
    const float* __restrict__ agg, const float* __restrict__ deg,
    const float* __restrict__ h,      // [N, INC]
    const float* __restrict__ root,   // [INC, 64]
    const float* __restrict__ bias,
    const float* __restrict__ gam, const float* __restrict__ bet,
    float* __restrict__ hout,                  // layer1 output [N,64]
    const int* __restrict__ batch, float* __restrict__ pool,  // layer2 pooling
    int nNodes)
{
    int wid = threadIdx.x >> 5, lane = threadIdx.x & 31;
    int nid = blockIdx.x * 8 + wid;
    if (nid >= nNodes) return;
    float invd = 1.0f / fmaxf(deg[nid], 1.0f);
    float v[2];
    #pragma unroll
    for (int q = 0; q < 2; ++q) {
        int o = lane + 32 * q;
        float acc = agg[nid * 64 + o] * invd + bias[o];
        for (int i = 0; i < INC; ++i) acc += h[nid * INC + i] * root[i * 64 + o];
        v[q] = fmaxf(acc, 0.0f);       // relu before LN
    }
    float s = v[0] + v[1];
    for (int off = 16; off >= 1; off >>= 1) s += __shfl_xor(s, off, 32);
    float mean = s * (1.0f / 64.0f);
    float d0 = v[0] - mean, d1 = v[1] - mean;
    float vs = d0 * d0 + d1 * d1;
    for (int off = 16; off >= 1; off >>= 1) vs += __shfl_xor(vs, off, 32);
    float rstd = rsqrtf(vs * (1.0f / 64.0f) + 1e-5f);
    #pragma unroll
    for (int q = 0; q < 2; ++q) {
        int o = lane + 32 * q;
        float outv = (v[q] - mean) * rstd * gam[o] + bet[o];
        if (FINAL) atomicAdd(&pool[batch[nid] * 64 + o], outv);
        else       hout[nid * 64 + o] = outv;
    }
}

// ---------------------------------------------------------------- driver
extern "C" void kernel_launch(void* const* d_in, const int* in_sizes, int n_in,
                              void* d_out, int out_size, void* d_ws, size_t ws_size,
                              hipStream_t stream) {
    const float* x     = (const float*)d_in[0];
    const int*   eidx  = (const int*)  d_in[1];
    const float* eattr = (const float*)d_in[2];
    const int*   batch = (const int*)  d_in[3];
    const float* eW1_1 = (const float*)d_in[4];
    const float* eb1_1 = (const float*)d_in[5];
    const float* eW2_1 = (const float*)d_in[6];
    const float* eb2_1 = (const float*)d_in[7];
    const float* root1 = (const float*)d_in[8];
    const float* bias1 = (const float*)d_in[9];
    const float* g1v   = (const float*)d_in[10];
    const float* b1v   = (const float*)d_in[11];
    const float* eW1_2 = (const float*)d_in[12];
    const float* eb1_2 = (const float*)d_in[13];
    const float* eW2_2 = (const float*)d_in[14];
    const float* eb2_2 = (const float*)d_in[15];
    const float* root2 = (const float*)d_in[16];
    const float* bias2 = (const float*)d_in[17];
    const float* g2v   = (const float*)d_in[18];
    const float* b2v   = (const float*)d_in[19];
    float* out = (float*)d_out;

    const int N = N_NODES, E = N_EDGES;

    // workspace layout (floats)
    float*  ws   = (float*)d_ws;
    float*  agg  = ws;                   // N*64
    float*  deg  = agg + (size_t)N * 64; // N
    float*  h1   = deg + N;              // N*64
    float*  hb   = h1 + (size_t)N * 64;  // N*64
    float*  pool = hb + (size_t)N * 64;  // 4096
    float*  cnt  = pool + 4096;          // 64
    __bf16* bp1  = (__bf16*)(cnt + 64);  // 256*64  = 16384 bf16
    __bf16* bp2  = bp1 + 16384;          // 8192*64 = 524288 bf16

    dim3 b256(256);
    auto gr = [](int n) { return dim3((n + 255) / 256); };

    // init accumulators
    zero_kernel<<<gr(N * 64), b256, 0, stream>>>(agg, N * 64);
    zero_kernel<<<gr(N),      b256, 0, stream>>>(deg, N);
    zero_kernel<<<gr(4096),   b256, 0, stream>>>(pool, 4096);
    zero_kernel<<<gr(64),     b256, 0, stream>>>(cnt, 64);

    deg_kernel<<<gr(E), b256, 0, stream>>>(eidx, deg, E);

    // prepack B' (bf16, WMMA fragment order)
    pack_b_kernel<64, 4>  <<<gr(64 * 4 * 64),   b256, 0, stream>>>(eW2_1, bp1);
    pack_b_kernel<128, 64><<<gr(128 * 64 * 64), b256, 0, stream>>>(eW2_2, bp2);

    // ---- layer 1 ----
    hb_kernel<4><<<dim3((N + 7) / 8), b256, 0, stream>>>(x, eb2_1, hb, N);
    edge_msg_kernel<64, 4><<<dim3((E + 15) / 16), dim3(32), 0, stream>>>(
        x, eidx, eattr, eW1_1, eb1_1, bp1, hb, agg, E);
    node_update_kernel<4, false><<<dim3((N + 7) / 8), b256, 0, stream>>>(
        agg, deg, x, root1, bias1, g1v, b1v, h1, nullptr, nullptr, N);

    // ---- layer 2 ----
    zero_kernel<<<gr(N * 64), b256, 0, stream>>>(agg, N * 64);
    hb_kernel<64><<<dim3((N + 7) / 8), b256, 0, stream>>>(h1, eb2_2, hb, N);
    edge_msg_kernel<128, 64><<<dim3((E + 15) / 16), dim3(32), 0, stream>>>(
        h1, eidx, eattr, eW1_2, eb1_2, bp2, hb, agg, E);
    node_update_kernel<64, true><<<dim3((N + 7) / 8), b256, 0, stream>>>(
        agg, deg, h1, root2, bias2, g2v, b2v, nullptr, batch, pool, N);

    // ---- global mean pool ----
    cnt_kernel<<<gr(N), b256, 0, stream>>>(batch, cnt, N);
    final_pool_kernel<<<gr(N_GRAPHS * HIDDEN), b256, 0, stream>>>(pool, cnt, out);
}
